// PretrainModel_89532888252748
// MI455X (gfx1250) — matrix operand order repaired and checked by reading, not compiled, not run on hardware
//
#include <hip/hip_runtime.h>
#include <hip/hip_bf16.h>
#include <math.h>

// ---------------------------------------------------------------------------
// MI455X (gfx1250) transformer encoder + pairwise scorer.
// All GEMMs via v_wmma_f32_16x16x32_bf16 (bf16 inputs, fp32 accumulate).
// Global->LDS staging uses GLOBAL_LOAD_ASYNC_TO_LDS_B128 (ASYNCcnt path)
// when the toolchain exposes the builtin, else synchronous vector copies.
// Workspace use: ~150 MB (assumed available in d_ws).
// ---------------------------------------------------------------------------

typedef __bf16 bf16;
typedef __attribute__((ext_vector_type(8)))  float v8f;
typedef __attribute__((ext_vector_type(16))) bf16  v16bf;
typedef __attribute__((ext_vector_type(8)))  bf16  v8bf;
typedef __attribute__((ext_vector_type(4)))  int   v4i;

#define NSEQ   512
#define BATCH  16
#define DMODEL 512
#define NHEAD  8
#define DHEAD  64
#define DFFN   2048
#define NLAYER 4
#define NCLS   3

#define LDSS 48  // LDS row stride (elements); rows stay 16B-aligned

#if defined(__has_builtin)
#if __has_builtin(__builtin_amdgcn_global_load_async_to_lds_b128)
#define HAVE_ASYNC_LDS 1
#endif
#if __has_builtin(__builtin_amdgcn_s_wait_asynccnt)
#define HAVE_WAIT_ASYNC 1
#endif
#endif

#ifdef HAVE_ASYNC_LDS
typedef __attribute__((address_space(1))) v4i* gv4i_p;   // global int4*
typedef __attribute__((address_space(3))) v4i* lv4i_p;   // LDS int4*
#endif

// 16-byte global -> LDS copy; async (ASYNCcnt) when available
__device__ __forceinline__ void g2lds16(const bf16* g, bf16* l) {
#ifdef HAVE_ASYNC_LDS
    __builtin_amdgcn_global_load_async_to_lds_b128(
        (gv4i_p)(void*)g, (lv4i_p)(void*)l, 0, 0);
#else
    *(v8bf*)l = *(const v8bf*)g;
#endif
}

__device__ __forceinline__ void wait_async_done() {
#ifdef HAVE_ASYNC_LDS
#ifdef HAVE_WAIT_ASYNC
    __builtin_amdgcn_s_wait_asynccnt(0);
#else
    asm volatile("s_wait_asynccnt 0" ::: "memory");
#endif
#endif
}

// ---------------------------------------------------------------------------
// WMMA fragment loads (wave32 layouts per CDNA5 ISA 7.12.2)
// A 16x32 bf16: lane<16 -> row=lane, K {0..7,16..23}; lane>=16 -> K {8..15,24..31}
// B 32x16 bf16 (stored N-major in LDS): lane%16 = column, lane/16 selects K half
// ---------------------------------------------------------------------------
__device__ __forceinline__ v16bf ldfragA(const bf16* As, int rowBase, int lane) {
    int row = rowBase + (lane & 15);
    int kb  = (lane >> 4) << 3;            // 0 or 8
    const bf16* p = As + row * LDSS + kb;
    v8bf lo = *(const v8bf*)p;
    v8bf hi = *(const v8bf*)(p + 16);
    v16bf r;
#pragma unroll
    for (int i = 0; i < 8; ++i) { r[i] = lo[i]; r[i + 8] = hi[i]; }
    return r;
}

__device__ __forceinline__ v16bf ldfragB(const bf16* Bs, int colBase, int lane) {
    int n  = colBase + (lane & 15);
    int kb = (lane >> 4) << 3;             // 0 or 8
    const bf16* p = Bs + n * LDSS + kb;
    v8bf lo = *(const v8bf*)p;
    v8bf hi = *(const v8bf*)(p + 16);
    v16bf r;
#pragma unroll
    for (int i = 0; i < 8; ++i) { r[i] = lo[i]; r[i + 8] = hi[i]; }
    return r;
}

// ---------------------------------------------------------------------------
// Generic batched bf16 WMMA GEMM: C = alpha*(A@B) [+bias] [relu] [+residual]
// A: [M,K] row-major (lda). B: [K,N] row-major (ldb) or transB -> B is [N,K].
// Requires M%256==0, N%64==0, K%32==0 (true for every call site here).
// Block 256 threads (8 wave32), block tile 256x64, wave tile 32x64
// -> 8 v_wmma per 32-wide K chunk per wave.
// ---------------------------------------------------------------------------
__global__ __launch_bounds__(256)
void gemm_bf16_wmma(const bf16* __restrict__ A, int lda, long long strideA,
                    const bf16* __restrict__ B, int ldb, long long strideB, int transB,
                    float* __restrict__ C32, bf16* __restrict__ Cbf,
                    int ldc, long long strideC,
                    int M, int N, int K,
                    const float* __restrict__ bias,
                    const float* __restrict__ residual,
                    float alpha, int relu) {
    __shared__ bf16 smem[(256 + 64) * LDSS];   // 30 KB
    bf16* As = smem;
    bf16* Bs = smem + 256 * LDSS;

    const int t    = threadIdx.x;
    const int lane = t & 31;
    const int wave = t >> 5;
    const int wm   = wave * 32;               // wave row offset in block tile

    const int rowBlock = blockIdx.y * 256;
    const int colBlock = blockIdx.x * 64;
    const int z        = blockIdx.z;

    A += (long long)z * strideA;
    B += (long long)z * strideB;
    const long long coff = (long long)z * strideC;

    v8f acc[2][4];
#pragma unroll
    for (int i = 0; i < 2; ++i)
#pragma unroll
        for (int j = 0; j < 4; ++j)
#pragma unroll
            for (int r = 0; r < 8; ++r) acc[i][j][r] = 0.0f;

    for (int kc = 0; kc < K; kc += 32) {
        // ---- stage A tile [256 x 32]: one row per thread, 4x16B ----
        {
            const bf16* src = A + (long long)(rowBlock + t) * lda + kc;
            bf16* dst = As + t * LDSS;
#pragma unroll
            for (int i = 0; i < 4; ++i) g2lds16(src + i * 8, dst + i * 8);
            if (kc + 32 < K)
                __builtin_prefetch(src + 32, 0, 1);   // global_prefetch_b8
        }
        // ---- stage B tile as Bs[n][kk] (N-major) ----
        if (transB) {
            int n  = t >> 2;               // 0..63
            int kb = (t & 3) * 8;          // 0,8,16,24
            const bf16* src = B + (long long)(colBlock + n) * ldb + kc + kb;
            g2lds16(src, Bs + n * LDSS + kb);
        } else {
            int kk = t & 31;               // 0..31
            int nb = (t >> 5) * 8;         // 0..56 step 8
            const bf16* src = B + (long long)(kc + kk) * ldb + colBlock + nb;
            v8bf v = *(const v8bf*)src;
#pragma unroll
            for (int i = 0; i < 8; ++i) Bs[(nb + i) * LDSS + kk] = v[i];
        }
        wait_async_done();
        __syncthreads();

        v16bf a0 = ldfragA(As, wm,      lane);
        v16bf a1 = ldfragA(As, wm + 16, lane);
        v16bf b0 = ldfragB(Bs,  0, lane);
        v16bf b1 = ldfragB(Bs, 16, lane);
        v16bf b2 = ldfragB(Bs, 32, lane);
        v16bf b3 = ldfragB(Bs, 48, lane);

        acc[0][0] = __builtin_amdgcn_wmma_f32_16x16x32_bf16(false, a0, false, b0, (short)0, acc[0][0], false, false);
        acc[0][1] = __builtin_amdgcn_wmma_f32_16x16x32_bf16(false, a0, false, b1, (short)0, acc[0][1], false, false);
        acc[0][2] = __builtin_amdgcn_wmma_f32_16x16x32_bf16(false, a0, false, b2, (short)0, acc[0][2], false, false);
        acc[0][3] = __builtin_amdgcn_wmma_f32_16x16x32_bf16(false, a0, false, b3, (short)0, acc[0][3], false, false);
        acc[1][0] = __builtin_amdgcn_wmma_f32_16x16x32_bf16(false, a1, false, b0, (short)0, acc[1][0], false, false);
        acc[1][1] = __builtin_amdgcn_wmma_f32_16x16x32_bf16(false, a1, false, b1, (short)0, acc[1][1], false, false);
        acc[1][2] = __builtin_amdgcn_wmma_f32_16x16x32_bf16(false, a1, false, b2, (short)0, acc[1][2], false, false);
        acc[1][3] = __builtin_amdgcn_wmma_f32_16x16x32_bf16(false, a1, false, b3, (short)0, acc[1][3], false, false);
        __syncthreads();
    }

    // ---- epilogue (C/D layout: VGPR r -> row r + (lane>=16 ? 8:0), col lane%16) ----
    const int crow0 = rowBlock + wm + ((lane >> 4) << 3);
    const int ccol0 = colBlock + (lane & 15);
#pragma unroll
    for (int tm = 0; tm < 2; ++tm)
#pragma unroll
        for (int tn = 0; tn < 4; ++tn) {
            int colg = ccol0 + tn * 16;
            float bv = bias ? bias[colg] : 0.0f;
#pragma unroll
            for (int r = 0; r < 8; ++r) {
                int rowg = crow0 + tm * 16 + r;
                float v = alpha * acc[tm][tn][r] + bv;
                if (relu) v = fmaxf(v, 0.0f);
                long long off = coff + (long long)rowg * ldc + colg;
                if (residual) v += residual[off];
                if (C32) C32[off] = v;
                if (Cbf) Cbf[off] = (bf16)v;
            }
        }
}

// ---------------------------------------------------------------------------
// Elementwise / reduction kernels
// ---------------------------------------------------------------------------
__global__ void convert_f32_bf16(const float* __restrict__ s, bf16* __restrict__ d, int n) {
    int i = blockIdx.x * blockDim.x + threadIdx.x;
    if (i < n) d[i] = (bf16)s[i];
}

// emb_table[src]*sqrt(D) + sinusoidal PE, write x[B,N,D]
__global__ void embed_kernel(const int* __restrict__ src, const float* __restrict__ table,
                             float* __restrict__ x) {
    int idx = blockIdx.x * blockDim.x + threadIdx.x;   // (b,n,d)
    if (idx >= BATCH * NSEQ * DMODEL) return;
    int d = idx & (DMODEL - 1);
    int n = (idx / DMODEL) & (NSEQ - 1);
    int b = idx / (DMODEL * NSEQ);
    int tok = src[n * BATCH + b];
    float freq = expf(-logf(10000.0f) * (float)(d & ~1) / (float)DMODEL);
    float ang  = (float)n * freq;
    float pe   = (d & 1) ? cosf(ang) : sinf(ang);
    x[idx] = table[(long long)tok * DMODEL + d] * 22.62741699796952f + pe;
}

// LayerNorm over D=512; one block (128 thr) per row; optional f32/bf16 outputs
__global__ void ln_kernel(const float* __restrict__ x,
                          const float* __restrict__ g, const float* __restrict__ bta,
                          float* __restrict__ o32, bf16* __restrict__ obf) {
    __shared__ float red[128];
    int row = blockIdx.x, t = threadIdx.x;
    const float* xr = x + (long long)row * DMODEL;
    float v[4], s = 0.0f;
#pragma unroll
    for (int i = 0; i < 4; ++i) { v[i] = xr[t + i * 128]; s += v[i]; }
    red[t] = s; __syncthreads();
    for (int o = 64; o > 0; o >>= 1) { if (t < o) red[t] += red[t + o]; __syncthreads(); }
    float mean = red[0] * (1.0f / DMODEL); __syncthreads();
    s = 0.0f;
#pragma unroll
    for (int i = 0; i < 4; ++i) { float d = v[i] - mean; s += d * d; }
    red[t] = s; __syncthreads();
    for (int o = 64; o > 0; o >>= 1) { if (t < o) red[t] += red[t + o]; __syncthreads(); }
    float rstd = rsqrtf(red[0] * (1.0f / DMODEL) + 1e-6f);
#pragma unroll
    for (int i = 0; i < 4; ++i) {
        int c = t + i * 128;
        float y = (v[i] - mean) * rstd * g[c] + bta[c];
        if (o32) o32[(long long)row * DMODEL + c] = y;
        if (obf) obf[(long long)row * DMODEL + c] = (bf16)y;
    }
}

// masked softmax over N=512 keys; one block (128 thr) per (h,q) row; bf16 probs
__global__ void softmax_kernel(const float* __restrict__ S, bf16* __restrict__ P,
                               const int* __restrict__ lengths, int bidx) {
    __shared__ float red[128];
    int row = blockIdx.x, t = threadIdx.x;
    int len = lengths[bidx];
    const float* sr = S + (long long)row * NSEQ;
    float v[4], m = -3.4e38f;
#pragma unroll
    for (int i = 0; i < 4; ++i) {
        int j = t + i * 128;
        float s = sr[j];
        if (j >= len) s = -1e9f;
        v[i] = s; m = fmaxf(m, s);
    }
    red[t] = m; __syncthreads();
    for (int o = 64; o > 0; o >>= 1) { if (t < o) red[t] = fmaxf(red[t], red[t + o]); __syncthreads(); }
    float mx = red[0]; __syncthreads();
    float sum = 0.0f;
#pragma unroll
    for (int i = 0; i < 4; ++i) { v[i] = __expf(v[i] - mx); sum += v[i]; }
    red[t] = sum; __syncthreads();
    for (int o = 64; o > 0; o >>= 1) { if (t < o) red[t] += red[t + o]; __syncthreads(); }
    float inv = 1.0f / red[0];
#pragma unroll
    for (int i = 0; i < 4; ++i)
        P[(long long)row * NSEQ + t + i * 128] = (bf16)(v[i] * inv);
}

// xf[B,N,D] -> out[N,B,D]
__global__ void transpose_kernel(const float* __restrict__ xf, float* __restrict__ out) {
    int idx = blockIdx.x * blockDim.x + threadIdx.x;   // (n,b,d)
    if (idx >= NSEQ * BATCH * DMODEL) return;
    int d = idx & (DMODEL - 1);
    int b = (idx / DMODEL) & (BATCH - 1);
    int n = idx / (DMODEL * BATCH);
    out[idx] = xf[((long long)b * NSEQ + n) * DMODEL + d];
}

// la = xf @ inf_W[:D], lb = xf @ inf_W[D:]  (C=3, per-row dot products)
__global__ void infproj_kernel(const float* __restrict__ xf, const float* __restrict__ W,
                               float* __restrict__ la, float* __restrict__ lb) {
    int row = blockIdx.x * blockDim.x + threadIdx.x;   // 0..B*N-1
    if (row >= BATCH * NSEQ) return;
    const float* xr = xf + (long long)row * DMODEL;
    float a0 = 0, a1 = 0, a2 = 0, b0 = 0, b1 = 0, b2 = 0;
    for (int d = 0; d < DMODEL; ++d) {
        float v = xr[d];
        const float* wa = W + d * NCLS;
        const float* wb = W + (DMODEL + d) * NCLS;
        a0 += v * wa[0]; a1 += v * wa[1]; a2 += v * wa[2];
        b0 += v * wb[0]; b1 += v * wb[1]; b2 += v * wb[2];
    }
    la[row * NCLS + 0] = a0; la[row * NCLS + 1] = a1; la[row * NCLS + 2] = a2;
    lb[row * NCLS + 0] = b0; lb[row * NCLS + 1] = b1; lb[row * NCLS + 2] = b2;
}

// log_softmax over C=3 of la[b,i,:]+lb[b,j,:], out[(i*N+j), b, :]
__global__ void logsoftmax_kernel(const float* __restrict__ la, const float* __restrict__ lb,
                                  float* __restrict__ out) {
    int idx = blockIdx.x * blockDim.x + threadIdx.x;   // ((i*N+j)*B + b)
    if (idx >= NSEQ * NSEQ * BATCH) return;
    int b  = idx & (BATCH - 1);
    int ij = idx / BATCH;
    int j  = ij & (NSEQ - 1);
    int i  = ij / NSEQ;
    const float* pa = la + ((long long)b * NSEQ + i) * NCLS;
    const float* pb = lb + ((long long)b * NSEQ + j) * NCLS;
    float l0 = pa[0] + pb[0], l1 = pa[1] + pb[1], l2 = pa[2] + pb[2];
    float m = fmaxf(l0, fmaxf(l1, l2));
    float lse = m + __logf(__expf(l0 - m) + __expf(l1 - m) + __expf(l2 - m));
    float* o = out + (long long)idx * NCLS;
    o[0] = l0 - lse; o[1] = l1 - lse; o[2] = l2 - lse;
}

// ---------------------------------------------------------------------------
extern "C" void kernel_launch(void* const* d_in, const int* in_sizes, int n_in,
                              void* d_out, int out_size, void* d_ws, size_t ws_size,
                              hipStream_t stream) {
    const int*   src     = (const int*)d_in[0];
    const int*   lengths = (const int*)d_in[1];
    const float* emb     = (const float*)d_in[2];
    const float* ln1_g   = (const float*)d_in[3];
    const float* ln1_b   = (const float*)d_in[4];
    const float* Wq      = (const float*)d_in[5];
    const float* bq      = (const float*)d_in[6];
    const float* Wk      = (const float*)d_in[7];
    const float* bk      = (const float*)d_in[8];
    const float* Wv      = (const float*)d_in[9];
    const float* bv      = (const float*)d_in[10];
    const float* Wo      = (const float*)d_in[11];
    const float* bo      = (const float*)d_in[12];
    const float* ln2_g   = (const float*)d_in[13];
    const float* ln2_b   = (const float*)d_in[14];
    const float* W1      = (const float*)d_in[15];
    const float* b1      = (const float*)d_in[16];
    const float* W2      = (const float*)d_in[17];
    const float* b2      = (const float*)d_in[18];
    const float* lnf_g   = (const float*)d_in[19];
    const float* lnf_b   = (const float*)d_in[20];
    const float* inf_W   = (const float*)d_in[21];

    float* out_attn = (float*)d_out;                               // [N,B,D]
    float* out_inf  = (float*)d_out + (long long)NSEQ * BATCH * DMODEL;

    const long long BND = (long long)BATCH * NSEQ * DMODEL;        // 4,194,304
    const long long HNN = (long long)NHEAD * NSEQ * NSEQ;          // 2,097,152

    // workspace arena
    char* p = (char*)d_ws;
    auto alloc = [&](size_t bytes) -> void* {
        void* r = (void*)p;
        p += (bytes + 255) & ~(size_t)255;
        return r;
    };
    float* x      = (float*)alloc(sizeof(float) * BND);
    float* xf     = (float*)alloc(sizeof(float) * BND);
    bf16*  h_bf   = (bf16*)alloc(sizeof(bf16) * BND);
    bf16*  q_bf   = (bf16*)alloc(sizeof(bf16) * BND);
    bf16*  k_bf   = (bf16*)alloc(sizeof(bf16) * BND);
    bf16*  v_bf   = (bf16*)alloc(sizeof(bf16) * BND);
    bf16*  ctx_bf = (bf16*)alloc(sizeof(bf16) * BND);
    float* scores = (float*)alloc(sizeof(float) * HNN);            // per-b
    bf16*  probs  = (bf16*)alloc(sizeof(bf16) * HNN);              // per-b
    bf16*  ffn_bf = (bf16*)alloc(sizeof(bf16) * (long long)BATCH * NSEQ * DFFN);
    float* la     = (float*)alloc(sizeof(float) * BATCH * NSEQ * NCLS);
    float* lb     = (float*)alloc(sizeof(float) * BATCH * NSEQ * NCLS);
    bf16*  Wq_bf  = (bf16*)alloc(sizeof(bf16) * (long long)NLAYER * DMODEL * DMODEL);
    bf16*  Wk_bf  = (bf16*)alloc(sizeof(bf16) * (long long)NLAYER * DMODEL * DMODEL);
    bf16*  Wv_bf  = (bf16*)alloc(sizeof(bf16) * (long long)NLAYER * DMODEL * DMODEL);
    bf16*  Wo_bf  = (bf16*)alloc(sizeof(bf16) * (long long)NLAYER * DMODEL * DMODEL);
    bf16*  W1_bf  = (bf16*)alloc(sizeof(bf16) * (long long)NLAYER * DMODEL * DFFN);
    bf16*  W2_bf  = (bf16*)alloc(sizeof(bf16) * (long long)NLAYER * DFFN * DMODEL);

    auto cvt = [&](const float* s, bf16* d, long long n) {
        convert_f32_bf16<<<(int)((n + 255) / 256), 256, 0, stream>>>(s, d, (int)n);
    };
    cvt(Wq, Wq_bf, (long long)NLAYER * DMODEL * DMODEL);
    cvt(Wk, Wk_bf, (long long)NLAYER * DMODEL * DMODEL);
    cvt(Wv, Wv_bf, (long long)NLAYER * DMODEL * DMODEL);
    cvt(Wo, Wo_bf, (long long)NLAYER * DMODEL * DMODEL);
    cvt(W1, W1_bf, (long long)NLAYER * DMODEL * DFFN);
    cvt(W2, W2_bf, (long long)NLAYER * DFFN * DMODEL);

    auto gemm = [&](const bf16* A, int lda, long long sA,
                    const bf16* Bm, int ldb, long long sB, int transB,
                    float* C32, bf16* Cbf, int ldc, long long sC,
                    int M, int N, int K, const float* bias, const float* res,
                    float alpha, int relu, int batch) {
        dim3 g(N / 64, M / 256, batch);
        gemm_bf16_wmma<<<g, 256, 0, stream>>>(A, lda, sA, Bm, ldb, sB, transB,
                                              C32, Cbf, ldc, sC, M, N, K,
                                              bias, res, alpha, relu);
    };

    const int ROWS = BATCH * NSEQ;   // 8192

    // embeddings + PE
    embed_kernel<<<(int)((BND + 255) / 256), 256, 0, stream>>>(src, emb, x);

    for (int l = 0; l < NLAYER; ++l) {
        const long long wDD = (long long)l * DMODEL * DMODEL;
        // h = LN1(x) (bf16)
        ln_kernel<<<ROWS, 128, 0, stream>>>(x, ln1_g + l * DMODEL, ln1_b + l * DMODEL,
                                            nullptr, h_bf);
        // q/k/v = h @ W + b  (bf16 out)
        gemm(h_bf, DMODEL, 0, Wq_bf + wDD, DMODEL, 0, 0, nullptr, q_bf, DMODEL, 0,
             ROWS, DMODEL, DMODEL, bq + l * DMODEL, nullptr, 1.0f, 0, 1);
        gemm(h_bf, DMODEL, 0, Wk_bf + wDD, DMODEL, 0, 0, nullptr, k_bf, DMODEL, 0,
             ROWS, DMODEL, DMODEL, bk + l * DMODEL, nullptr, 1.0f, 0, 1);
        gemm(h_bf, DMODEL, 0, Wv_bf + wDD, DMODEL, 0, 0, nullptr, v_bf, DMODEL, 0,
             ROWS, DMODEL, DMODEL, bv + l * DMODEL, nullptr, 1.0f, 0, 1);

        // attention, one batch element at a time (heads batched via grid.z)
        for (int b = 0; b < BATCH; ++b) {
            const bf16* qb = q_bf + (long long)b * NSEQ * DMODEL;
            const bf16* kb = k_bf + (long long)b * NSEQ * DMODEL;
            const bf16* vb = v_bf + (long long)b * NSEQ * DMODEL;
            // S[h] = Q_h @ K_h^T / sqrt(dh)
            gemm(qb, DMODEL, DHEAD, kb, DMODEL, DHEAD, 1,
                 scores, nullptr, NSEQ, (long long)NSEQ * NSEQ,
                 NSEQ, NSEQ, DHEAD, nullptr, nullptr, 0.125f, 0, NHEAD);
            softmax_kernel<<<NHEAD * NSEQ, 128, 0, stream>>>(scores, probs, lengths, b);
            // ctx[h] = P_h @ V_h  (bf16 out, packed back into [N, D] at col h*64)
            gemm(probs, NSEQ, (long long)NSEQ * NSEQ, vb, DMODEL, DHEAD, 0,
                 nullptr, ctx_bf + (long long)b * NSEQ * DMODEL, DMODEL, DHEAD,
                 NSEQ, DHEAD, NSEQ, nullptr, nullptr, 1.0f, 0, NHEAD);
        }
        // x = x + ctx @ Wo + bo (in-place residual)
        gemm(ctx_bf, DMODEL, 0, Wo_bf + wDD, DMODEL, 0, 0, x, nullptr, DMODEL, 0,
             ROWS, DMODEL, DMODEL, bo + l * DMODEL, x, 1.0f, 0, 1);

        // FFN
        ln_kernel<<<ROWS, 128, 0, stream>>>(x, ln2_g + l * DMODEL, ln2_b + l * DMODEL,
                                            nullptr, h_bf);
        gemm(h_bf, DMODEL, 0, W1_bf + (long long)l * DMODEL * DFFN, DFFN, 0, 0,
             nullptr, ffn_bf, DFFN, 0,
             ROWS, DFFN, DMODEL, b1 + l * DFFN, nullptr, 1.0f, 1, 1);
        gemm(ffn_bf, DFFN, 0, W2_bf + (long long)l * DFFN * DMODEL, DMODEL, 0, 0,
             x, nullptr, DMODEL, 0,
             ROWS, DMODEL, DFFN, b2 + l * DMODEL, x, 1.0f, 0, 1);
    }

    // final LN, outputs
    ln_kernel<<<ROWS, 128, 0, stream>>>(x, lnf_g, lnf_b, xf, nullptr);
    transpose_kernel<<<(int)((BND + 255) / 256), 256, 0, stream>>>(xf, out_attn);
    infproj_kernel<<<(ROWS + 255) / 256, 256, 0, stream>>>(xf, inf_W, la, lb);
    logsoftmax_kernel<<<(NSEQ * NSEQ * BATCH + 255) / 256, 256, 0, stream>>>(la, lb, out_inf);
}
